// AttentionLayer_53197464929114
// MI455X (gfx1250) — compile-verified, matrix operand order checked by reading, not run on hardware
//
#include <hip/hip_runtime.h>
#include <hip/hip_bf16.h>

#define Bn 4096
#define Sn 200
#define En 64
#define Hn 64
#define SP 208            // S padded to 13 * 16
#define XCOLS 136         // 128 bf16 + 8 pad -> 272 B row stride (bank-conflict free)

typedef __attribute__((ext_vector_type(16))) __bf16  v16bf;
typedef __attribute__((ext_vector_type(8)))  float   v8f;
typedef __attribute__((ext_vector_type(4)))  unsigned int uint4v;
typedef __attribute__((ext_vector_type(2)))  unsigned int uint2v;

union Frag {
    uint4v q[2];
    v16bf  v;
};

__device__ __forceinline__ unsigned int f2bf(float f) {
    unsigned int u = __float_as_uint(f);
    u += 0x7FFFu + ((u >> 16) & 1u);      // round-to-nearest-even
    return u >> 16;
}

// group-of-32 ds_swizzle, xor mode: offset = (xor<<10) | and_mask(0x1F)
#define SWZ(x, imm) \
    __int_as_float(__builtin_amdgcn_ds_swizzle(__float_as_int(x), imm))
#define XOR1  0x041F
#define XOR2  0x081F
#define XOR4  0x101F
#define XOR8  0x201F
#define XOR16 0x401F

// ---------------------------------------------------------------------------
// Prep 1: QA[b][h] = q[b] . (W1_q + W1_d) + b1[h]   (fp32, tiny)
// ---------------------------------------------------------------------------
__global__ __launch_bounds__(256) void prep_qa(const float* __restrict__ q,
                                               const float* __restrict__ W1,
                                               const float* __restrict__ b1,
                                               float* __restrict__ QA) {
    int idx = blockIdx.x * 256 + threadIdx.x;          // B*H threads
    if (idx >= Bn * Hn) return;
    int b = idx >> 6, h = idx & 63;
    const float* qb = q + (size_t)b * En;
    float acc = b1[h];
#pragma unroll 8
    for (int e = 0; e < En; ++e)
        acc += qb[e] * (W1[e * Hn + h] + W1[(128 + e) * Hn + h]);
    QA[idx] = acc;
}

// ---------------------------------------------------------------------------
// Prep 2: WT[n][k] (bf16, transposed): rows 0..63 = W1_k - W1_d, 64..127 = W1_m
// ---------------------------------------------------------------------------
__global__ __launch_bounds__(256) void prep_wt(const float* __restrict__ W1,
                                               unsigned short* __restrict__ WT) {
    int idx = blockIdx.x * 256 + threadIdx.x;          // 64*128 threads
    if (idx >= Hn * 128) return;
    int n = idx >> 7, k = idx & 127;
    float v;
    if (k < 64) v = W1[(64 + k) * Hn + n] - W1[(128 + k) * Hn + n];
    else        v = W1[(192 + (k - 64)) * Hn + n];
    WT[idx] = (unsigned short)f2bf(v);
}

// ---------------------------------------------------------------------------
// Main: one workgroup per batch element. 256 threads = 8 waves.
// ---------------------------------------------------------------------------
__global__ __launch_bounds__(256) void attn_main(
    const float* __restrict__ q,   const float* __restrict__ K,
    const float* __restrict__ V,   const int*   __restrict__ mask,
    const float* __restrict__ QA,  const unsigned short* __restrict__ WT,
    const float* __restrict__ W2,  const float* __restrict__ b2,
    float* __restrict__ out)
{
    __shared__ unsigned short Xs[SP * XCOLS];   // [K | q*K] bf16, padded rows
    __shared__ unsigned short Ws[Hn * XCOLS];   // WT bf16, padded rows
    __shared__ float qsh[En], qash[Hn], w2sh[Hn];
    __shared__ float scoresP[4 * SP];           // per-N-tile score slices
    __shared__ float wgt[SP];
    __shared__ float red[512];
    __shared__ float b2sh;

    const int    t     = threadIdx.x;
    const int    b     = blockIdx.x;
    const size_t kbase = (size_t)b * Sn * En;

    // Warm L2 with V while the MLP GEMM runs (global_prefetch_b8)
    __builtin_prefetch(V + kbase + (size_t)t * 50, 0, 1);

    // ---- stage per-batch vectors + weight matrix into LDS -----------------
    if (t < En) {
        qsh[t]  = q [(size_t)b * En + t];
        qash[t] = QA[(size_t)b * Hn + t];
        w2sh[t] = W2[t];
    }
    if (t == 0) b2sh = b2[0];
    {
        const uint4v* WT4 = (const uint4v*)WT;          // 8 ushorts per chunk
#pragma unroll
        for (int idx = t; idx < (Hn * 128) / 8; idx += 256) {
            int n = idx >> 4, c8 = (idx & 15) * 8;
            *(uint4v*)&Ws[n * XCOLS + c8] = WT4[idx];
        }
    }
    __syncthreads();

    // ---- build X = [bf16(K) | bf16(q*K)] in LDS (float4-granular) ---------
    {
        const float4* K4 = (const float4*)(K + kbase);
        for (int idx = t; idx < (Sn * En) / 4; idx += 256) {
            float4 kv = K4[idx];
            int row = idx >> 4;                 // 16 float4 per 64-wide row
            int e4  = (idx & 15) * 4;
            uint2v pk, pkq;
            pk.x  = f2bf(kv.x) | (f2bf(kv.y) << 16);
            pk.y  = f2bf(kv.z) | (f2bf(kv.w) << 16);
            pkq.x = f2bf(qsh[e4]     * kv.x) | (f2bf(qsh[e4 + 1] * kv.y) << 16);
            pkq.y = f2bf(qsh[e4 + 2] * kv.z) | (f2bf(qsh[e4 + 3] * kv.w) << 16);
            *(uint2v*)&Xs[row * XCOLS + e4]      = pk;
            *(uint2v*)&Xs[row * XCOLS + 64 + e4] = pkq;
        }
        for (int idx = t; idx < 8 * 64; idx += 256) {   // zero pad rows 200..207
            int row = Sn + (idx >> 6), c2 = (idx & 63) * 2;
            *(unsigned int*)&Xs[row * XCOLS + c2] = 0u;
        }
    }
    __syncthreads();

    // ---- GEMM: scores_h[S,64] = X[S,128] @ W[128,64] via bf16 WMMA --------
    const int wave  = __builtin_amdgcn_readfirstlane(t) >> 5;   // scalar
    const int lane  = t & 31;
    const int nt    = wave & 3;                       // N-tile for this wave
    const int mt0   = (wave < 4) ? 0 : 7;             // M-tile range split
    const int mt1   = (wave < 4) ? 7 : 13;
    const int l15   = lane & 15;
    const int lhalf = lane >> 4;
    const int boff  = lhalf * 16;                     // K-half byte offset

    Frag bfrag[4];
    {
        const char* wrow = (const char*)&Ws[(nt * 16 + l15) * XCOLS];
#pragma unroll
        for (int ks = 0; ks < 4; ++ks) {
            bfrag[ks].q[0] = *(const uint4v*)(wrow + ks * 64 + boff);
            bfrag[ks].q[1] = *(const uint4v*)(wrow + ks * 64 + boff + 32);
        }
    }
    const float qa_n = qash[nt * 16 + l15];
    const float w2_n = w2sh[nt * 16 + l15];
    const bool  hb3  = (l15 & 8) != 0;
    const bool  hb2  = (l15 & 4) != 0;
    const bool  hb1  = (l15 & 2) != 0;
    const int   slot = (l15 >> 1) & 7;                // final row slot per lane

    for (int mt = mt0; mt < mt1; ++mt) {
        const char* arow = (const char*)&Xs[(mt * 16 + l15) * XCOLS];
        Frag a0, a1, a2, a3;                 // all 8 b128 loads issued up front
        a0.q[0] = *(const uint4v*)(arow + boff);
        a0.q[1] = *(const uint4v*)(arow + boff + 32);
        a1.q[0] = *(const uint4v*)(arow + 64 + boff);
        a1.q[1] = *(const uint4v*)(arow + 64 + boff + 32);
        a2.q[0] = *(const uint4v*)(arow + 128 + boff);
        a2.q[1] = *(const uint4v*)(arow + 128 + boff + 32);
        a3.q[0] = *(const uint4v*)(arow + 192 + boff);
        a3.q[1] = *(const uint4v*)(arow + 192 + boff + 32);
        v8f acc = {};
        acc = __builtin_amdgcn_wmma_f32_16x16x32_bf16(false, a0.v, false, bfrag[0].v, (short)0, acc, false, false);
        acc = __builtin_amdgcn_wmma_f32_16x16x32_bf16(false, a1.v, false, bfrag[1].v, (short)0, acc, false, false);
        acc = __builtin_amdgcn_wmma_f32_16x16x32_bf16(false, a2.v, false, bfrag[2].v, (short)0, acc, false, false);
        acc = __builtin_amdgcn_wmma_f32_16x16x32_bf16(false, a3.v, false, bfrag[3].v, (short)0, acc, false, false);

        // Fused epilogue: p_i = relu(acc_i + QA_n) * W2_n, then reduce the
        // 16 columns (lanes) for 8 rows at once with a slot-folding
        // butterfly: 8 -> 4 -> 2 -> 1 slots, 8 swizzles total.
        float p[8];
#pragma unroll
        for (int i = 0; i < 8; ++i)
            p[i] = fmaxf(acc[i] + qa_n, 0.f) * w2_n;

        float s0, s1, s2, s3, u0, u1, vv;
        {   // xor 8: keep slots (b3?4..7:0..3)
            float x0 = hb3 ? p[0] : p[4];
            float x1 = hb3 ? p[1] : p[5];
            float x2 = hb3 ? p[2] : p[6];
            float x3 = hb3 ? p[3] : p[7];
            float r0 = SWZ(x0, XOR8), r1 = SWZ(x1, XOR8);
            float r2 = SWZ(x2, XOR8), r3 = SWZ(x3, XOR8);
            s0 = (hb3 ? p[4] : p[0]) + r0;
            s1 = (hb3 ? p[5] : p[1]) + r1;
            s2 = (hb3 ? p[6] : p[2]) + r2;
            s3 = (hb3 ? p[7] : p[3]) + r3;
        }
        {   // xor 4
            float x0 = hb2 ? s0 : s2;
            float x1 = hb2 ? s1 : s3;
            float r0 = SWZ(x0, XOR4), r1 = SWZ(x1, XOR4);
            u0 = (hb2 ? s2 : s0) + r0;
            u1 = (hb2 ? s3 : s1) + r1;
        }
        {   // xor 2
            float x = hb1 ? u0 : u1;
            float r = SWZ(x, XOR2);
            vv = (hb1 ? u1 : u0) + r;
        }
        vv += SWZ(vv, XOR1);                           // xor 1
        // lane pair (l, l^1) now holds row slot = (l15>>1)&7; even lanes store
        if ((lane & 1) == 0)
            scoresP[nt * SP + mt * 16 + (lhalf << 3) + slot] = vv;
    }
    __syncthreads();

    // ---- masked softmax over S: wave-level swizzle + 8-entry combine ------
    const bool valid = (t < Sn);
    float sc = -3.0e38f;
    if (valid) {
        sc = b2sh + scoresP[t] + scoresP[SP + t] +
             scoresP[2 * SP + t] + scoresP[3 * SP + t];
        if (mask[(size_t)b * Sn + t] == 0) sc = -1.0e9f;
    }
    float m = sc;
    m = fmaxf(m, SWZ(m, XOR1));
    m = fmaxf(m, SWZ(m, XOR2));
    m = fmaxf(m, SWZ(m, XOR4));
    m = fmaxf(m, SWZ(m, XOR8));
    m = fmaxf(m, SWZ(m, XOR16));
    if (lane == 0) red[wave] = m;
    __syncthreads();
    float mx = red[0];
#pragma unroll
    for (int i = 1; i < 8; ++i) mx = fmaxf(mx, red[i]);

    const float ex = valid ? __expf(sc - mx) : 0.f;
    float sum = ex;
    sum += SWZ(sum, XOR1);
    sum += SWZ(sum, XOR2);
    sum += SWZ(sum, XOR4);
    sum += SWZ(sum, XOR8);
    sum += SWZ(sum, XOR16);
    __syncthreads();                          // red[0..7] reads done
    if (lane == 0) red[wave] = sum;
    __syncthreads();
    float tot = red[0];
#pragma unroll
    for (int i = 1; i < 8; ++i) tot += red[i];

    const float w = ex * (1.0f / tot);
    if (valid) {
        wgt[t] = w;
        out[(size_t)Bn * En + (size_t)b * Sn + t] = w;   // weights output
    }
    __syncthreads();                          // wgt ready; red free for reuse

    // ---- output[b] = sum_s w_s * V[b,s,:]  (float2-granular) --------------
    {
        const int e2   = (t & 31) * 2;                   // element pair
        const int part = t >> 5;                         // 8 partials of 25 s
        const float2* vb2 = (const float2*)(V + kbase);
        float ax = 0.f, ay = 0.f;
#pragma unroll 5
        for (int i = 0; i < 25; ++i) {
            int s = part * 25 + i;
            float2 vv = vb2[s * 32 + (e2 >> 1)];
            float ws = wgt[s];
            ax += ws * vv.x;
            ay += ws * vv.y;
        }
        red[part * 64 + e2]     = ax;
        red[part * 64 + e2 + 1] = ay;
        __syncthreads();
        if (t < 64) {
            float o = 0.f;
#pragma unroll
            for (int p = 0; p < 8; ++p) o += red[p * 64 + t];
            out[(size_t)b * En + t] = o;
        }
    }
}

// ---------------------------------------------------------------------------
extern "C" void kernel_launch(void* const* d_in, const int* in_sizes, int n_in,
                              void* d_out, int out_size, void* d_ws, size_t ws_size,
                              hipStream_t stream) {
    (void)in_sizes; (void)n_in; (void)out_size; (void)ws_size;
    const float* query  = (const float*)d_in[0];
    const float* keys   = (const float*)d_in[1];
    const float* values = (const float*)d_in[2];
    const int*   mask   = (const int*)  d_in[3];
    const float* W1     = (const float*)d_in[4];
    const float* b1     = (const float*)d_in[5];
    const float* W2     = (const float*)d_in[6];
    const float* b2     = (const float*)d_in[7];
    float* out = (float*)d_out;

    float*          QA = (float*)d_ws;                                  // 1 MB
    unsigned short* WT = (unsigned short*)((char*)d_ws +
                                           (size_t)Bn * Hn * sizeof(float));

    prep_qa <<<(Bn * Hn + 255) / 256, 256, 0, stream>>>(query, W1, b1, QA);
    prep_wt <<<(Hn * 128 + 255) / 256, 256, 0, stream>>>(W1, WT);
    attn_main<<<Bn, 256, 0, stream>>>(query, keys, values, mask,
                                      QA, WT, W2, b2, out);
}
